// EP_GAT_PP_64493228917300
// MI455X (gfx1250) — matrix operand order repaired and checked by reading, not compiled, not run on hardware
//
#include <hip/hip_runtime.h>

// ---------------------------------------------------------------------------
// GAT edge-softmax aggregation for MI455X (gfx1250, wave32, WMMA)
//
// el = e_ft @ W        -> f16 WMMA 16x16x32, f32 accumulate (bandwidth-bound:
//                         154 MB @ 23.3 TB/s needs ~1 PFLOP/s -> WMMA only).
//                         64-row M tiles: 4x B-fragment reuse, 4x less W L2
//                         re-read than 16-row tiles. A-fragments batched so
//                         ds_load latency overlaps WMMA issue.
// e  = <el[src],el[dst]> per head, leakyrelu, edge-softmax over dst segments
// out = mean_h( ft * asum + bias )   where asum[n,h] = sum of attention
//                                     weights over incoming edges (scalar!)
// ---------------------------------------------------------------------------

typedef __attribute__((ext_vector_type(16))) _Float16 v16h;
typedef __attribute__((ext_vector_type(8)))  float    v8f;

#define GAT_H    4
#define GAT_D    32
#define GAT_FOUT 128   // H*D
#define GAT_FIN  256
#define GAT_NEG  0.2f

// ---------------- Kernel 1: el = e_ft @ W via WMMA -------------------------
// Block = 256 threads = 8 waves. Block covers a 64-row M tile x all 128 cols.
// Wave w owns output cols [w*16, w*16+16) and all 4 M-subtiles (16 rows each).
// K=256 processed in 2 chunks of 128 through LDS.
__global__ __launch_bounds__(256) void gat_gemm_wmma(
    const float* __restrict__ eft,   // [N,256]
    const float* __restrict__ W,     // [256,128]
    float* __restrict__ el,          // [N,128]
    int N)
{
    constexpr int MT  = 64;                      // M tile rows per block
    constexpr int KC  = 128;                     // K chunk
    constexpr int LDP = KC + 8;                  // pad to spread LDS banks
    __shared__ _Float16 Alds[MT][LDP];           // ~17.4 KB
    __shared__ _Float16 Blds[8][16][LDP];        // ~34.8 KB

    const int tid   = threadIdx.x;
    const int wave  = tid >> 5;
    const int lane  = tid & 31;
    const int mbase = blockIdx.x * MT;
    const int m     = lane & 15;                 // A row-in-subtile / D col
    const int khalf = lane >> 4;                 // low/high 16 lanes
    const bool full = (mbase + MT) <= N;

    v8f acc[4];
    #pragma unroll
    for (int mt = 0; mt < 4; ++mt)
        acc[mt] = (v8f){0.f,0.f,0.f,0.f,0.f,0.f,0.f,0.f};

    for (int kc = 0; kc < GAT_FIN; kc += KC) {
        // prefetch next A chunk while this one is loaded/consumed
        if (kc + KC < GAT_FIN) {
            int pr = mbase + (tid >> 2);         // 64 rows, 4 threads/row
            if (pr < N)
                __builtin_prefetch(eft + (size_t)pr * GAT_FIN + kc + KC
                                       + (tid & 3) * 32, 0, 1);
        }
        // cooperative A tile load: 64 x 128, f32 -> f16
        for (int i = tid; i < MT * KC; i += 256) {
            int r = i >> 7;                      // /KC
            int c = i & (KC - 1);
            int row = mbase + r;
            float v = (full || row < N) ? eft[(size_t)row * GAT_FIN + kc + c]
                                        : 0.f;
            Alds[r][c] = (_Float16)v;
        }
        // per-wave B slice load (transpose to [col][k])
        for (int k = lane; k < KC; k += 32) {
            const float* wp = W + (size_t)(kc + k) * GAT_FOUT + wave * 16;
            #pragma unroll
            for (int c = 0; c < 16; ++c)
                Blds[wave][c][k] = (_Float16)wp[c];
        }
        __syncthreads();

        // 4 WMMA K-steps per chunk, 4 M-subtiles each (B fragment reused 4x).
        // A 16x32 f16 layout: lanes0-15 (M=lane): V0-3=K0..7,  V4-7=K16..23
        //                     lanes16-31        : V0-3=K8..15, V4-7=K24..31
        // B 32x16 f16 layout: same K order as A (lanes0-15: K0..7,K16..23; lanes16-31: K8..15,K24..31)
        // All 4 A fragments are loaded before the 4 WMMAs so the ds_load
        // latency is overlapped instead of a dscnt-0 wait per WMMA.
        #pragma unroll
        for (int step = 0; step < KC / 32; ++step) {
            const int k0 = step * 32;
            v16h b;
            #pragma unroll
            for (int j = 0; j < 16; ++j)
                b[j] = Blds[wave][m][k0 + khalf * 8 + (j & 7) + (j >> 3) * 16];
            v16h a[4];
            #pragma unroll
            for (int mt = 0; mt < 4; ++mt) {
                #pragma unroll
                for (int j = 0; j < 8; ++j)
                    a[mt][j]     = Alds[mt * 16 + m][k0 + khalf * 8 + j];
                #pragma unroll
                for (int j = 0; j < 8; ++j)
                    a[mt][8 + j] = Alds[mt * 16 + m][k0 + 16 + khalf * 8 + j];
            }
            #pragma unroll
            for (int mt = 0; mt < 4; ++mt)
                acc[mt] = __builtin_amdgcn_wmma_f32_16x16x32_f16(
                              false, a[mt], false, b, (short)0, acc[mt],
                              false, false);
        }
        __syncthreads();
    }

    // D layout: VGPR r -> row M = r (lanes0-15) / r+8 (lanes16-31), col = m
    const int colbase = wave * 16 + m;
    if (full) {
        #pragma unroll
        for (int mt = 0; mt < 4; ++mt)
            #pragma unroll
            for (int r = 0; r < 8; ++r)
                el[(size_t)(mbase + mt * 16 + r + khalf * 8) * GAT_FOUT
                   + colbase] = acc[mt][r];
    } else {
        #pragma unroll
        for (int mt = 0; mt < 4; ++mt)
            #pragma unroll
            for (int r = 0; r < 8; ++r) {
                int row = mbase + mt * 16 + r + khalf * 8;
                if (row < N)
                    el[(size_t)row * GAT_FOUT + colbase] = acc[mt][r];
            }
    }
}

// ---------------- float atomic max (monotonic int mapping) -----------------
__device__ __forceinline__ void atomic_max_f32(float* addr, float v) {
    if (v >= 0.f)
        atomicMax((int*)addr, __float_as_int(v));
    else
        atomicMin((unsigned int*)addr, __float_as_uint(v));
}

// ---------------- Kernel 0: init reduction buffers -------------------------
__global__ __launch_bounds__(256) void gat_init(float* __restrict__ emax,
                                                float* __restrict__ esum,
                                                float* __restrict__ asum,
                                                int nh)
{
    int i = blockIdx.x * blockDim.x + threadIdx.x;
    if (i < nh) {
        emax[i] = -__builtin_huge_valf();
        esum[i] = 0.f;
        asum[i] = 0.f;
    }
}

// ---------------- Kernel 2: edge scores + segment max ----------------------
__global__ __launch_bounds__(256) void gat_edge_score(
    const float* __restrict__ el,
    const int* __restrict__ src,
    const int* __restrict__ dst,
    float* __restrict__ e,       // [E,H]
    float* __restrict__ emax,    // [N,H]
    int E)
{
    int idx = blockIdx.x * blockDim.x + threadIdx.x;
    if (idx >= E * GAT_H) return;
    int edge = idx >> 2;
    int h    = idx & 3;
    int s = src[edge], d = dst[edge];
    const float4* ps = (const float4*)(el + (size_t)s * GAT_FOUT + h * GAT_D);
    const float4* pd = (const float4*)(el + (size_t)d * GAT_FOUT + h * GAT_D);
    float acc = 0.f;
    #pragma unroll
    for (int j = 0; j < GAT_D / 4; ++j) {
        float4 a = ps[j], b = pd[j];
        acc += a.x * b.x + a.y * b.y + a.z * b.z + a.w * b.w;
    }
    acc = (acc > 0.f) ? acc : GAT_NEG * acc;
    e[idx] = acc;
    atomic_max_f32(&emax[(size_t)d * GAT_H + h], acc);
}

// ---------------- Kernel 3: exp + segment sum ------------------------------
__global__ __launch_bounds__(256) void gat_exp(
    const int* __restrict__ dst,
    float* __restrict__ e,           // in: score, out: exp(score - max)
    const float* __restrict__ emax,
    float* __restrict__ esum,
    int E)
{
    int idx = blockIdx.x * blockDim.x + threadIdx.x;
    if (idx >= E * GAT_H) return;
    int edge = idx >> 2;
    int h    = idx & 3;
    int d = dst[edge];
    float ex = __expf(e[idx] - emax[(size_t)d * GAT_H + h]);
    e[idx] = ex;
    atomicAdd(&esum[(size_t)d * GAT_H + h], ex);
}

// ---------------- Kernel 4: attention-weight segment sum -------------------
// Because the message is ft[dst]*a summed onto dst, only sum_a per (node,head)
// is needed — scalar atomics instead of 32-float vector atomics (32x less).
__global__ __launch_bounds__(256) void gat_asum(
    const int* __restrict__ dst,
    const float* __restrict__ e,     // exp values
    const float* __restrict__ esum,
    float* __restrict__ asum,
    int E)
{
    int idx = blockIdx.x * blockDim.x + threadIdx.x;
    if (idx >= E * GAT_H) return;
    int edge = idx >> 2;
    int h    = idx & 3;
    int d = dst[edge];
    float a = e[idx] / esum[(size_t)d * GAT_H + h];
    atomicAdd(&asum[(size_t)d * GAT_H + h], a);
}

// ---------------- Kernel 5: node-level output ------------------------------
__global__ __launch_bounds__(256) void gat_out(
    const float* __restrict__ ft,    // [N,H,D]
    const float* __restrict__ asum,  // [N,H]
    const float* __restrict__ bias,  // [H*D]
    float* __restrict__ out,         // [N,D]
    int N)
{
    int idx = blockIdx.x * blockDim.x + threadIdx.x;
    if (idx >= N * GAT_D) return;
    int n = idx >> 5;                // D = 32
    int d = idx & 31;
    float s = 0.f;
    #pragma unroll
    for (int h = 0; h < GAT_H; ++h)
        s += ft[(size_t)n * GAT_FOUT + h * GAT_D + d] * asum[(size_t)n * GAT_H + h]
           + bias[h * GAT_D + d];
    out[idx] = s * (1.0f / GAT_H);
}

// ---------------------------------------------------------------------------
extern "C" void kernel_launch(void* const* d_in, const int* in_sizes, int n_in,
                              void* d_out, int out_size, void* d_ws, size_t ws_size,
                              hipStream_t stream)
{
    const float* ft   = (const float*)d_in[0];   // [N,H,D]
    const float* eft  = (const float*)d_in[1];   // [N,256]
    const float* W    = (const float*)d_in[2];   // [256,128]
    const float* bias = (const float*)d_in[3];   // [128]
    const int*   src  = (const int*)d_in[4];     // [E]
    const int*   dst  = (const int*)d_in[5];     // [E]

    const int N = in_sizes[0] / (GAT_H * GAT_D);
    const int E = in_sizes[4];

    // workspace layout (floats): el | e | emax | esum | asum  (~82 MB total)
    float* el   = (float*)d_ws;
    float* e    = el   + (size_t)N * GAT_FOUT;
    float* emax = e    + (size_t)E * GAT_H;
    float* esum = emax + (size_t)N * GAT_H;
    float* asum = esum + (size_t)N * GAT_H;

    const int nh = N * GAT_H;
    const int eh = E * GAT_H;

    gat_init<<<(nh + 255) / 256, 256, 0, stream>>>(emax, esum, asum, nh);
    gat_gemm_wmma<<<(N + 63) / 64, 256, 0, stream>>>(eft, W, el, N);
    gat_edge_score<<<(eh + 255) / 256, 256, 0, stream>>>(el, src, dst, e, emax, E);
    gat_exp<<<(eh + 255) / 256, 256, 0, stream>>>(dst, e, emax, esum, E);
    gat_asum<<<(eh + 255) / 256, 256, 0, stream>>>(dst, e, esum, asum, E);
    gat_out<<<(N * GAT_D + 255) / 256, 256, 0, stream>>>(ft, asum, bias,
                                                         (float*)d_out, N);
}
